// CREStereoBase_188978561113
// MI455X (gfx1250) — matrix operand order repaired and checked by reading, not compiled, not run on hardware
//
#include <hip/hip_runtime.h>
#include <hip/hip_bf16.h>

// ---------------------------------------------------------------------------
// CDNA5 (gfx1250) CREStereo forward — full pipeline.
// All GEMM-like work (convs, attention projections, MLPs, GRU convs) runs on
// one WMMA implicit-GEMM kernel: wave32, 256-thread blocks (8 waves),
// block tile 128x64, K-tile 32, v_wmma_f32_16x16x32_f16, f32 accumulate.
// LDS tiles are stored fragment-major (ISA 7.12.2 lane mapping inverted at
// staging time) so fragment loads are 2x ds_load_b128 per lane.
// ---------------------------------------------------------------------------

typedef __attribute__((ext_vector_type(16))) _Float16 v16h;
typedef __attribute__((ext_vector_type(8)))  _Float16 v8h;
typedef __attribute__((ext_vector_type(8)))  float    v8f;

#define DEV __device__ __forceinline__

#define ACT_NONE     0
#define ACT_RELU     1
#define ACT_TANH     2
#define ACT_SIGSCALE 3   // (sigmoid(x)-0.5)*2
#define ACT_SIGMOID  4
#define ACT_ELU1     5   // elu(x)+1
#define ACT_QUARTER  6   // 0.25*x

DEV float apply_act(float v, int act) {
  switch (act) {
    case ACT_RELU:     return v > 0.f ? v : 0.f;
    case ACT_TANH:     return tanhf(v);
    case ACT_SIGSCALE: return (1.f / (1.f + __expf(-v)) - 0.5f) * 2.f;
    case ACT_SIGMOID:  return 1.f / (1.f + __expf(-v));
    case ACT_ELU1:     return v > 0.f ? v + 1.f : __expf(v);
    case ACT_QUARTER:  return 0.25f * v;
  }
  return v;
}

// fragment-major lane slot: 16 halves padded to 24 (48B stride -> 2-phase b128)
#define FRAG_STRIDE 24
#define FRAG_GROUP  (32 * FRAG_STRIDE)   // one 16x32 (A) or 32x16 (B) subtile

DEV v16h frag_ld(const _Float16* p) {
  int lane = threadIdx.x & 31;
  const v8h* q = (const v8h*)(p + lane * FRAG_STRIDE);
  v8h lo = q[0], hi = q[1];
  return __builtin_shufflevector(lo, hi, 0, 1, 2, 3, 4, 5, 6, 7,
                                 8, 9, 10, 11, 12, 13, 14, 15);
}

DEV v8f wmma_f16(v16h a, v16h b, v8f c) {
  return __builtin_amdgcn_wmma_f32_16x16x32_f16(false, a, false, b,
                                                (short)0, c, false, false);
}

// ---- implicit GEMM conv ----------------------------------------------------
#define BM 128
#define BN 64
#define BK 32

__global__ void __launch_bounds__(256)
k_conv_wmma(const _Float16* __restrict__ W, const _Float16* __restrict__ X,
            const float* __restrict__ bias,
            float* __restrict__ Y, _Float16* __restrict__ Y16,
            int M, int K, int Nb, int Cin, int H, int Wi,
            int Ho, int Wo, int KH, int KW, int stride, int padH, int padW,
            int act)
{
  // 8 A-subtiles (16 rows each), 4 B-subtiles (16 cols each), fragment-major
  __shared__ __align__(16) _Float16 sAf[8 * FRAG_GROUP];
  __shared__ __align__(16) _Float16 sBf[4 * FRAG_GROUP];

  const int tid  = threadIdx.x;
  const int wave = tid >> 5;
  const int wm   = wave >> 1;
  const int wn   = wave & 1;

  const int m0 = blockIdx.y * BM;
  const int n0 = blockIdx.x * BN;
  const long Ntot = (long)Nb * Ho * Wo;
  const int  khw  = KH * KW;
  const int  howo = Ho * Wo;

  v8f c00{}, c01{}, c10{}, c11{};

  for (int k0 = 0; k0 < K; k0 += BK) {
    // stage A (weights) into fragment-major layout: 128x32 halves, 16/thread
#pragma unroll 4
    for (int t = tid; t < BM * BK; t += 256) {
      int r = t >> 5, k = t & 31;
      int gm = m0 + r, gk = k0 + k;
      _Float16 v = (gm < M && gk < K) ? W[(long)gm * K + gk] : (_Float16)0;
      // invert ISA A-layout: lane = ((k>>3)&1)*16 + (r&15);
      // i = 2*( (k>>4)*4 + ((k&7)>>1) ) + (k&1)
      int tr   = r >> 4;
      int lane = (((k >> 3) & 1) << 4) + (r & 15);
      int i    = (((k >> 4) << 3) + ((k & 7) & ~1)) + (k & 1);
      sAf[(tr * 32 + lane) * FRAG_STRIDE + i] = v;
    }
    if (k0 + BK < K) {
      int pr = m0 + (tid >> 1);
      if (pr < M) __builtin_prefetch(&W[(long)pr * K + k0 + BK], 0, 1);
    }
    // stage B (im2col gather) into fragment-major layout: 32x64, 8/thread
#pragma unroll 4
    for (int t = tid; t < BK * BN; t += 256) {
      int k = t >> 6, c = t & 63;
      int gk = k0 + k;
      long gj = n0 + c;
      _Float16 v = (_Float16)0;
      if (gk < K && gj < Ntot) {
        int ci  = gk / khw;
        int rem = gk - ci * khw;
        int fy  = rem / KW, fx = rem - fy * KW;
        int n   = (int)(gj / howo);
        int rj  = (int)(gj - (long)n * howo);
        int oy  = rj / Wo, ox = rj - oy * Wo;
        int iy  = oy * stride - padH + fy;
        int ix  = ox * stride - padW + fx;
        if (iy >= 0 && iy < H && ix >= 0 && ix < Wi)
          v = X[(((long)n * Cin + ci) * H + iy) * Wi + ix];
      }
      // invert ISA B-layout: lane = ((k>>3)&1)*16 + (c&15); i = (k>>4)*8 + (k&7)
      int cg   = c >> 4;
      int lane = (((k >> 3) & 1) << 4) + (c & 15);
      sBf[(cg * 32 + lane) * FRAG_STRIDE + (((k >> 4) << 3) + (k & 7))] = v;
    }
    __syncthreads();

    v16h a0 = frag_ld(&sAf[(wm * 2 + 0) * FRAG_GROUP]);
    v16h a1 = frag_ld(&sAf[(wm * 2 + 1) * FRAG_GROUP]);
    v16h b0 = frag_ld(&sBf[(wn * 2 + 0) * FRAG_GROUP]);
    v16h b1 = frag_ld(&sBf[(wn * 2 + 1) * FRAG_GROUP]);

    c00 = wmma_f16(a0, b0, c00);
    c01 = wmma_f16(a0, b1, c01);
    c10 = wmma_f16(a1, b0, c10);
    c11 = wmma_f16(a1, b1, c11);
    __syncthreads();
  }

  const int lane = threadIdx.x & 31;
  const int colL = lane & 15;
  const int rowH = 8 * (lane >> 4);

  const v8f* accs[4] = { &c00, &c01, &c10, &c11 };
#pragma unroll
  for (int s = 0; s < 4; ++s) {
    int gm_base = m0 + wm * 32 + ((s >> 1) << 4) + rowH;
    long gc = (long)n0 + wn * 32 + ((s & 1) << 4) + colL;
    if (gc >= Ntot) continue;
    int n  = (int)(gc / howo);
    int rj = (int)(gc - (long)n * howo);
    int oy = rj / Wo, ox = rj - oy * Wo;
    const v8f& cc = *accs[s];
#pragma unroll
    for (int r = 0; r < 8; ++r) {
      int gm = gm_base + r;
      if (gm < M) {
        float v = cc[r] + (bias ? bias[gm] : 0.f);
        v = apply_act(v, act);
        long idx = (((long)n * M + gm) * Ho + oy) * Wo + ox;
        if (Y)   Y[idx] = v;
        if (Y16) Y16[idx] = (_Float16)v;
      }
    }
  }
}

// ---- elementwise / util kernels -------------------------------------------

__global__ void k_f32_to_f16(const float* __restrict__ in,
                             _Float16* __restrict__ out, long n) {
  long i = (long)blockIdx.x * blockDim.x + threadIdx.x;
  if (i < n) out[i] = (_Float16)in[i];
}

__global__ void k_zero_f32(float* __restrict__ out, long n) {
  long i = (long)blockIdx.x * blockDim.x + threadIdx.x;
  if (i < n) out[i] = 0.f;
}

__global__ void k_add_inplace(float* __restrict__ a, const float* __restrict__ b,
                              long n) {
  long i = (long)blockIdx.x * blockDim.x + threadIdx.x;
  if (i < n) a[i] += b[i];
}

__global__ void k_gru_update(float* __restrict__ h, const float* __restrict__ z,
                             const float* __restrict__ q, long n) {
  long i = (long)blockIdx.x * blockDim.x + threadIdx.x;
  if (i < n) h[i] = (1.f - z[i]) * h[i] + z[i] * q[i];
}

__global__ void k_pack_f32h(const float* __restrict__ src, int sC, int sC0,
                            _Float16* __restrict__ dst, int dC, int dC0,
                            int Cc, int N, int HW) {
  long total = (long)N * Cc * HW;
  long i = (long)blockIdx.x * blockDim.x + threadIdx.x;
  if (i >= total) return;
  int hw = (int)(i % HW);
  long t = i / HW;
  int c = (int)(t % Cc);
  int n = (int)(t / Cc);
  dst[((long)n * dC + dC0 + c) * HW + hw] =
      (_Float16)src[((long)n * sC + sC0 + c) * HW + hw];
}

__global__ void k_pack_hh(const _Float16* __restrict__ src, int sC, int sC0,
                          _Float16* __restrict__ dst, int dC, int dC0,
                          int Cc, int N, int HW) {
  long total = (long)N * Cc * HW;
  long i = (long)blockIdx.x * blockDim.x + threadIdx.x;
  if (i >= total) return;
  int hw = (int)(i % HW);
  long t = i / HW;
  int c = (int)(t % Cc);
  int n = (int)(t / Cc);
  dst[((long)n * dC + dC0 + c) * HW + hw] =
      src[((long)n * sC + sC0 + c) * HW + hw];
}

__global__ void k_pack_rh(const float* __restrict__ r, const float* __restrict__ h,
                          _Float16* __restrict__ dst, int N, int HW) {
  long total = (long)N * 128 * HW;
  long i = (long)blockIdx.x * blockDim.x + threadIdx.x;
  if (i >= total) return;
  int hw = (int)(i % HW);
  long t = i / HW;
  int c = (int)(t % 128);
  int n = (int)(t / 128);
  long s = ((long)n * 128 + c) * HW + hw;
  dst[((long)n * 384 + c) * HW + hw] = (_Float16)(r[s] * h[s]);
}

__global__ void __launch_bounds__(256)
k_instnorm_relu(const float* __restrict__ in, _Float16* __restrict__ out,
                int HW) {
  const float* p = in  + (long)blockIdx.x * HW;
  _Float16*    q = out + (long)blockIdx.x * HW;
  __shared__ float red[2][8];
  float s = 0.f, ss = 0.f;
  for (int i = threadIdx.x; i < HW; i += 256) {
    float v = p[i];
    s += v; ss += v * v;
  }
#pragma unroll
  for (int off = 16; off > 0; off >>= 1) {
    s  += __shfl_down(s,  off, 32);
    ss += __shfl_down(ss, off, 32);
  }
  int wave = threadIdx.x >> 5, lane = threadIdx.x & 31;
  if (lane == 0) { red[0][wave] = s; red[1][wave] = ss; }
  __syncthreads();
  if (threadIdx.x == 0) {
    float S = 0.f, SS = 0.f;
    for (int w = 0; w < 8; ++w) { S += red[0][w]; SS += red[1][w]; }
    float m = S / HW;
    float var = SS / HW - m * m;
    red[0][0] = m;
    red[1][0] = rsqrtf(var + 1e-5f);
  }
  __syncthreads();
  float m = red[0][0], inv = red[1][0];
  for (int i = threadIdx.x; i < HW; i += 256) {
    float v = (p[i] - m) * inv;
    q[i] = (_Float16)(v > 0.f ? v : 0.f);
  }
}

__global__ void k_avgpool(const float* __restrict__ in,
                          float* __restrict__ out32,
                          _Float16* __restrict__ out16,
                          int NC, int H, int W, int k) {
  int Ho = H / k, Wo = W / k;
  long total = (long)NC * Ho * Wo;
  long i = (long)blockIdx.x * blockDim.x + threadIdx.x;
  if (i >= total) return;
  int wo = (int)(i % Wo);
  long t = i / Wo;
  int ho = (int)(t % Ho);
  int nc = (int)(t / Ho);
  const float* p = in + (long)nc * H * W;
  float s = 0.f;
  for (int dy = 0; dy < k; ++dy)
    for (int dx = 0; dx < k; ++dx)
      s += p[(ho * k + dy) * W + wo * k + dx];
  s *= 1.f / (k * k);
  if (out32) out32[i] = s;
  if (out16) out16[i] = (_Float16)s;
}

__global__ void k_net_inp(const float* __restrict__ fmap,
                          float* __restrict__ net, float* __restrict__ inp,
                          int N, int HW) {
  long total = (long)N * 128 * HW;
  long i = (long)blockIdx.x * blockDim.x + threadIdx.x;
  if (i >= total) return;
  int hw = (int)(i % HW);
  long t = i / HW;
  int c = (int)(t % 128);
  int n = (int)(t / 128);
  const float* base = fmap + (long)n * 256 * HW;
  net[i] = tanhf(base[(long)c * HW + hw]);
  float v = base[(long)(c + 128) * HW + hw];
  inp[i] = v > 0.f ? v : 0.f;
}

__global__ void k_posenc(const float* __restrict__ in, float* __restrict__ out32,
                         _Float16* __restrict__ out16, int N, int H, int W) {
  const int C = 256;
  long total = (long)N * C * H * W;
  long i = (long)blockIdx.x * blockDim.x + threadIdx.x;
  if (i >= total) return;
  int w = (int)(i % W);
  long t = i / W;
  int h = (int)(t % H); t /= H;
  int c = (int)(t % C);
  int k = c >> 2, m = c & 3;
  float div = __expf(-(float)k * 0.14391156517f);   // ln(1e4)/64
  float pos = (m < 2) ? (float)(w + 1) : (float)(h + 1);
  float a = pos * div;
  float pe = (m & 1) ? __cosf(a) : __sinf(a);
  float v = in[i] + pe;
  out32[i] = v;
  out16[i] = (_Float16)v;
}

// ---- linear attention pieces (heads=8, d=32, C=256) ------------------------

__global__ void __launch_bounds__(256)
k_kv_ksum(const float* __restrict__ Kf, const float* __restrict__ Vf,
          float* __restrict__ KV, float* __restrict__ Ksum, int L) {
  int nh = blockIdx.x;
  int n = nh >> 3, h = nh & 7;
  const float* Kb = Kf + ((long)n * 256 + h * 32) * L;
  const float* Vb = Vf + ((long)n * 256 + h * 32) * L;
  float invL = 1.f / (float)L;
  for (int t = threadIdx.x; t < 1024; t += 256) {
    int d = t >> 5, v = t & 31;
    const float* kr = Kb + (long)d * L;
    const float* vr = Vb + (long)v * L;
    float s = 0.f;
    for (int l = 0; l < L; ++l) s += kr[l] * vr[l];
    KV[((long)nh * 32 + d) * 32 + v] = s * invL;
  }
  if (threadIdx.x < 32) {
    const float* kr = Kb + (long)threadIdx.x * L;
    float s = 0.f;
    for (int l = 0; l < L; ++l) s += kr[l];
    Ksum[(long)nh * 32 + threadIdx.x] = s;
  }
}

__global__ void k_zcoef(const float* __restrict__ Qf, const float* __restrict__ Ksum,
                        float* __restrict__ Z, int N, int L) {
  long total = (long)N * 8 * L;
  long i = (long)blockIdx.x * blockDim.x + threadIdx.x;
  if (i >= total) return;
  int l = (int)(i % L);
  int h = (int)((i / L) % 8);
  int n = (int)(i / ((long)8 * L));
  const float* Qb = Qf + ((long)n * 256 + h * 32) * L + l;
  const float* Ks = Ksum + ((long)n * 8 + h) * 32;
  float s = 0.f;
  for (int d = 0; d < 32; ++d) s += Qb[(long)d * L] * Ks[d];
  Z[i] = 1.f / (s + 1e-6f);
}

__global__ void k_msg(const float* __restrict__ Qf, const float* __restrict__ KV,
                      const float* __restrict__ Z, _Float16* __restrict__ msgh,
                      int N, int L) {
  long total = (long)N * 256 * L;
  long i = (long)blockIdx.x * blockDim.x + threadIdx.x;
  if (i >= total) return;
  int l = (int)(i % L);
  int c = (int)((i / L) % 256);
  int n = (int)(i / ((long)256 * L));
  int h = c >> 5, v = c & 31;
  const float* Qb = Qf + ((long)n * 256 + h * 32) * L + l;
  const float* kv = KV + (((long)n * 8 + h) * 32) * 32 + v;
  float s = 0.f;
  for (int d = 0; d < 32; ++d) s += Qb[(long)d * L] * kv[d * 32];
  float z = Z[((long)n * 8 + h) * L + l];
  msgh[i] = (_Float16)(s * z * (float)L);
}

__global__ void k_layernorm_fm(const float* __restrict__ in,
                               const float* __restrict__ g,
                               const float* __restrict__ b,
                               const float* __restrict__ residual,
                               float* __restrict__ out32,
                               _Float16* __restrict__ out16,
                               int N, int C, int L) {
  long total = (long)N * L;
  long i = (long)blockIdx.x * blockDim.x + threadIdx.x;
  if (i >= total) return;
  int l = (int)(i % L);
  int n = (int)(i / L);
  const float* p = in + (long)n * C * L + l;
  float s = 0.f, ss = 0.f;
  for (int c = 0; c < C; ++c) {
    float v = p[(long)c * L];
    s += v; ss += v * v;
  }
  float m = s / C;
  float inv = rsqrtf(ss / C - m * m + 1e-5f);
  for (int c = 0; c < C; ++c) {
    float v = (p[(long)c * L] - m) * inv * g[c] + b[c];
    if (residual) v += residual[((long)n * C + c) * L + l];
    long o = ((long)n * C + c) * L + l;
    if (out32) out32[o] = v;
    if (out16) out16[o] = (_Float16)v;
  }
}

// ---- correlation kernels ---------------------------------------------------

__global__ void k_corr_offset(const float* __restrict__ left,
                              const float* __restrict__ right,
                              const float* __restrict__ flow,
                              const float* __restrict__ extra,
                              _Float16* __restrict__ corr,
                              int N, int H, int W, int rx, int ry) {
  int HW = H * W;
  long total = (long)N * 9 * HW;
  long i = (long)blockIdx.x * blockDim.x + threadIdx.x;
  if (i >= total) return;
  int w = (int)(i % W);
  long t = i / W;
  int h = (int)(t % H); t /= H;
  int k = (int)(t % 9);
  int n = (int)(t / 9);
  int nx = 2 * rx + 1;
  float dx = (float)(k % nx - rx);
  float dy = (float)(k / nx - ry);
  long fb = (long)n * 2 * HW + (long)h * W + w;
  long eb = (long)n * 18 * HW + (long)h * W + w;
  float cx = (float)w + flow[fb]      + extra[eb + (long)(2 * k) * HW]     + dx;
  float cy = (float)h + flow[fb + HW] + extra[eb + (long)(2 * k + 1) * HW] + dy;
  float x0 = floorf(cx), y0 = floorf(cy);
  float wx = cx - x0, wy = cy - y0;
  int ix0 = (int)x0, iy0 = (int)y0, ix1 = ix0 + 1, iy1 = iy0 + 1;
  bool b00 = (ix0 >= 0 && ix0 < W && iy0 >= 0 && iy0 < H);
  bool b01 = (ix0 >= 0 && ix0 < W && iy1 >= 0 && iy1 < H);
  bool b10 = (ix1 >= 0 && ix1 < W && iy0 >= 0 && iy0 < H);
  bool b11 = (ix1 >= 0 && ix1 < W && iy1 >= 0 && iy1 < H);
  float w00 = (1 - wx) * (1 - wy), w01 = (1 - wx) * wy;
  float w10 = wx * (1 - wy),       w11 = wx * wy;
  int o00 = iy0 * W + ix0, o01 = iy1 * W + ix0;
  int o10 = iy0 * W + ix1, o11 = iy1 * W + ix1;
  const float* R  = right + (long)n * 256 * HW;
  const float* Lp = left  + (long)n * 256 * HW + (long)h * W + w;
  float acc[4] = {0.f, 0.f, 0.f, 0.f};
  for (int c = 0; c < 256; ++c) {
    const float* Rc = R + (long)c * HW;
    float v = 0.f;
    if (b00) v += w00 * Rc[o00];
    if (b01) v += w01 * Rc[o01];
    if (b10) v += w10 * Rc[o10];
    if (b11) v += w11 * Rc[o11];
    acc[c >> 6] += Lp[(long)c * HW] * v;
  }
  long ob = (long)n * 36 * HW + (long)h * W + w;
#pragma unroll
  for (int g = 0; g < 4; ++g)
    corr[ob + (long)(g * 9 + k) * HW] = (_Float16)(acc[g] * (1.f / 64.f));
}

__global__ void k_bilin_all(const float* __restrict__ right,
                            const float* __restrict__ flow,
                            float* __restrict__ rs, int N, int H, int W) {
  int HW = H * W;
  long total = (long)N * 256 * HW;
  long i = (long)blockIdx.x * blockDim.x + threadIdx.x;
  if (i >= total) return;
  int w = (int)(i % W);
  long t = i / W;
  int h = (int)(t % H); t /= H;
  int c = (int)(t % 256);
  int n = (int)(t / 256);
  long fb = (long)n * 2 * HW + (long)h * W + w;
  float cx = (float)w + flow[fb];
  float cy = (float)h + flow[fb + HW];
  float x0 = floorf(cx), y0 = floorf(cy);
  float wx = cx - x0, wy = cy - y0;
  int ix0 = (int)x0, iy0 = (int)y0, ix1 = ix0 + 1, iy1 = iy0 + 1;
  const float* Rc = right + ((long)n * 256 + c) * HW;
  float v = 0.f;
  if (ix0 >= 0 && ix0 < W && iy0 >= 0 && iy0 < H) v += (1 - wx) * (1 - wy) * Rc[iy0 * W + ix0];
  if (ix0 >= 0 && ix0 < W && iy1 >= 0 && iy1 < H) v += (1 - wx) * wy       * Rc[iy1 * W + ix0];
  if (ix1 >= 0 && ix1 < W && iy0 >= 0 && iy0 < H) v += wx * (1 - wy)       * Rc[iy0 * W + ix1];
  if (ix1 >= 0 && ix1 < W && iy1 >= 0 && iy1 < H) v += wx * wy             * Rc[iy1 * W + ix1];
  rs[i] = v;
}

__global__ void k_corr_shift(const float* __restrict__ left,
                             const float* __restrict__ rs,
                             _Float16* __restrict__ corr,
                             int N, int H, int W, int rx, int ry) {
  int HW = H * W;
  long total = (long)N * 9 * HW;
  long i = (long)blockIdx.x * blockDim.x + threadIdx.x;
  if (i >= total) return;
  int w = (int)(i % W);
  long t = i / W;
  int h = (int)(t % H); t /= H;
  int k = (int)(t % 9);
  int n = (int)(t / 9);
  int nx = 2 * rx + 1;
  int dx = k % nx - rx;
  int dy = k / nx - ry;
  int sy = min(max(h + dy, 0), H - 1);
  int sx = min(max(w + dx, 0), W - 1);
  const float* Rb = rs   + (long)n * 256 * HW + (long)sy * W + sx;
  const float* Lb = left + (long)n * 256 * HW + (long)h * W + w;
  float acc[4] = {0.f, 0.f, 0.f, 0.f};
  for (int c = 0; c < 256; ++c)
    acc[c >> 6] += Lb[(long)c * HW] * Rb[(long)c * HW];
  long ob = (long)n * 36 * HW + (long)h * W + w;
#pragma unroll
  for (int g = 0; g < 4; ++g)
    corr[ob + (long)(g * 9 + k) * HW] = (_Float16)(acc[g] * (1.f / 64.f));
}

// ---- convex upsample (rate 8) and align-corners resize ---------------------

__global__ void k_upsample(const float* __restrict__ flow,
                           const float* __restrict__ mask,
                           float* __restrict__ out, int N, int H, int W) {
  int HW = H * W;
  long total = (long)N * HW * 64;
  long i = (long)blockIdx.x * blockDim.x + threadIdx.x;
  if (i >= total) return;
  int rj = (int)(i & 7);
  int ri = (int)((i >> 3) & 7);
  long t = i >> 6;
  int w = (int)(t % W); t /= W;
  int h = (int)(t % H);
  int n = (int)(t / H);
  const float* mb = mask + (long)n * 576 * HW + (long)h * W + w;
  float e[9], mx = -1e30f;
#pragma unroll
  for (int k = 0; k < 9; ++k) {
    e[k] = mb[(long)(k * 64 + ri * 8 + rj) * HW];
    mx = fmaxf(mx, e[k]);
  }
  float se = 0.f;
#pragma unroll
  for (int k = 0; k < 9; ++k) { e[k] = __expf(e[k] - mx); se += e[k]; }
  float inv = 1.f / se;
  const float* fb = flow + (long)n * 2 * HW;
#pragma unroll
  for (int ch = 0; ch < 2; ++ch) {
    float s = 0.f;
#pragma unroll
    for (int k = 0; k < 9; ++k) {
      int yy = h + k / 3 - 1;
      int xx = w + k % 3 - 1;
      if (yy >= 0 && yy < H && xx >= 0 && xx < W)
        s += e[k] * 8.f * fb[(long)ch * HW + yy * W + xx];
    }
    out[(((long)n * 2 + ch) * (H * 8) + (long)h * 8 + ri) * (W * 8)
        + (long)w * 8 + rj] = s * inv;
  }
}

__global__ void k_interp(const float* __restrict__ in, float* __restrict__ out,
                         int N, int C, int H, int W, int Ho, int Wo,
                         float scale) {
  long total = (long)N * C * Ho * Wo;
  long i = (long)blockIdx.x * blockDim.x + threadIdx.x;
  if (i >= total) return;
  int wo = (int)(i % Wo);
  long t = i / Wo;
  int ho = (int)(t % Ho); t /= Ho;
  int c  = (int)(t % C);
  int n  = (int)(t / C);
  float y = (Ho > 1) ? (float)ho * (float)(H - 1) / (float)(Ho - 1) : 0.f;
  float x = (Wo > 1) ? (float)wo * (float)(W - 1) / (float)(Wo - 1) : 0.f;
  int y0 = (int)floorf(y), x0 = (int)floorf(x);
  int y1 = min(y0 + 1, H - 1), x1 = min(x0 + 1, W - 1);
  float wy = y - y0, wx = x - x0;
  const float* p = in + ((long)n * C + c) * H * W;
  float v = (p[y0 * W + x0] * (1 - wx) + p[y0 * W + x1] * wx) * (1 - wy)
          + (p[y1 * W + x0] * (1 - wx) + p[y1 * W + x1] * wx) * wy;
  out[i] = scale * v;
}

// ---------------------------------------------------------------------------
// Host orchestration
// ---------------------------------------------------------------------------

static inline long cdiv(long a, long b) { return (a + b - 1) / b; }

extern "C" void kernel_launch(void* const* d_in, const int* in_sizes, int n_in,
                              void* d_out, int out_size, void* d_ws, size_t ws_size,
                              hipStream_t stream) {
  (void)in_sizes; (void)n_in; (void)ws_size; (void)out_size;

  const float* frame1 = (const float*)d_in[0];
  const float* frame2 = (const float*)d_in[1];
  const float* fb1 = (const float*)d_in[12];
  const float* fb2 = (const float*)d_in[13];
  const float* fb3 = (const float*)d_in[14];
  const float* fb4 = (const float*)d_in[15];
  const float* fw1 = (const float*)d_in[16];
  const float* fw2 = (const float*)d_in[17];
  const float* fw3 = (const float*)d_in[18];
  const float* fw4 = (const float*)d_in[19];
  const float* off16_b = (const float*)d_in[20];
  const float* off16_w = (const float*)d_in[21];
  const float* off8_b  = (const float*)d_in[22];
  const float* off8_w  = (const float*)d_in[23];

  const int NB = 2;
  const int H0 = 480, W0 = 640;
  const int H1 = 240, W1 = 320;
  const int H2 = 120, W2 = 160;
  const int H3 = 60,  W3 = 80;
  const int H8 = 30,  W8 = 40;
  const int H16 = 15, W16 = 20;
  const int L16 = H16 * W16;
  const int HWmax = H3 * W3;

  char*  ws  = (char*)d_ws;
  size_t off = 0;
  auto alloc  = [&](size_t bytes) -> void* {
    off = (off + 255) & ~(size_t)255;
    void* p = ws + off;
    off += bytes;
    return p;
  };
  auto allocF = [&](long n) -> float*    { return (float*)   alloc((size_t)n * 4); };
  auto allocH = [&](long n) -> _Float16* { return (_Float16*)alloc((size_t)n * 2); };

  auto cvt = [&](const float* src, _Float16* dst, long n) {
    k_f32_to_f16<<<dim3((unsigned)cdiv(n, 256)), 256, 0, stream>>>(src, dst, n);
  };
  auto conv = [&](const _Float16* Wt, const _Float16* X, const float* bias,
                  float* Y, _Float16* Y16, int M, int Cin, int KH, int KW,
                  int H, int Wd, int stride, int padH, int padW, int act) {
    int Ho = (H + 2 * padH - KH) / stride + 1;
    int Wo = (Wd + 2 * padW - KW) / stride + 1;
    int K  = Cin * KH * KW;
    long Ntot = (long)NB * Ho * Wo;
    dim3 g((unsigned)cdiv(Ntot, BN), (unsigned)cdiv(M, BM));
    k_conv_wmma<<<g, 256, 0, stream>>>(Wt, X, bias, Y, Y16, M, K, NB, Cin,
                                       H, Wd, Ho, Wo, KH, KW, stride,
                                       padH, padW, act);
  };
  auto packF = [&](const float* s, int sC, int sC0, _Float16* d, int dC,
                   int dC0, int Cc, int HW) {
    long n = (long)NB * Cc * HW;
    k_pack_f32h<<<dim3((unsigned)cdiv(n, 256)), 256, 0, stream>>>(
        s, sC, sC0, d, dC, dC0, Cc, NB, HW);
  };
  auto packH = [&](const _Float16* s, int sC, int sC0, _Float16* d, int dC,
                   int dC0, int Cc, int HW) {
    long n = (long)NB * Cc * HW;
    k_pack_hh<<<dim3((unsigned)cdiv(n, 256)), 256, 0, stream>>>(
        s, sC, sC0, d, dC, dC0, Cc, NB, HW);
  };

  _Float16* w1h = allocH(64L * 3 * 49);
  _Float16* w2h = allocH(128L * 64 * 9);
  _Float16* w3h = allocH(192L * 128 * 9);
  _Float16* w4h = allocH(256L * 192);
  _Float16* o8wh  = allocH(18L * 256 * 9);
  _Float16* o16wh = allocH(18L * 256 * 9);

  struct AttW {
    _Float16 *qw, *kw, *vw, *mw, *m1w, *m2w;
    const float *ln1g, *ln1b, *ln2g, *ln2b;
  };
  AttW att[2];   // 0=self, 1=cross
  for (int a = 0; a < 2; ++a) {
    int base = a ? 2 : 24;
    att[a].qw  = allocH(65536);  att[a].kw  = allocH(65536);
    att[a].vw  = allocH(65536);  att[a].mw  = allocH(65536);
    att[a].m1w = allocH(262144); att[a].m2w = allocH(131072);
    cvt((const float*)d_in[base + 8], att[a].qw, 65536);
    cvt((const float*)d_in[base + 0], att[a].kw, 65536);
    cvt((const float*)d_in[base + 9], att[a].vw, 65536);
    cvt((const float*)d_in[base + 5], att[a].mw, 65536);
    cvt((const float*)d_in[base + 6], att[a].m1w, 262144);
    cvt((const float*)d_in[base + 7], att[a].m2w, 131072);
    att[a].ln1b = (const float*)d_in[base + 1];
    att[a].ln1g = (const float*)d_in[base + 2];
    att[a].ln2b = (const float*)d_in[base + 3];
    att[a].ln2g = (const float*)d_in[base + 4];
  }

  const float* u_conv_b   = (const float*)d_in[34];
  const float* u_convc1_b = (const float*)d_in[36];
  const float* u_convc2_b = (const float*)d_in[38];
  const float* u_convf1_b = (const float*)d_in[40];
  const float* u_convf2_b = (const float*)d_in[42];
  const float* u_q1_b     = (const float*)d_in[44];
  const float* u_q2_b     = (const float*)d_in[46];
  const float* u_r1_b     = (const float*)d_in[48];
  const float* u_r2_b     = (const float*)d_in[50];
  const float* u_z1_b     = (const float*)d_in[52];
  const float* u_z2_b     = (const float*)d_in[54];
  const float* u_fh1_b    = (const float*)d_in[56];
  const float* u_fh2_b    = (const float*)d_in[58];
  const float* u_mask1_b  = (const float*)d_in[60];
  const float* u_mask2_b  = (const float*)d_in[62];

  _Float16* u_conv_w   = allocH(126L * 256 * 9);
  _Float16* u_convc1_w = allocH(256L * 36);
  _Float16* u_convc2_w = allocH(192L * 256 * 9);
  _Float16* u_convf1_w = allocH(128L * 2 * 49);
  _Float16* u_convf2_w = allocH(64L * 128 * 9);
  _Float16* u_q1_w = allocH(128L * 384 * 5);
  _Float16* u_q2_w = allocH(128L * 384 * 5);
  _Float16* u_r1_w = allocH(128L * 384 * 5);
  _Float16* u_r2_w = allocH(128L * 384 * 5);
  _Float16* u_z1_w = allocH(128L * 384 * 5);
  _Float16* u_z2_w = allocH(128L * 384 * 5);
  _Float16* u_fh1_w   = allocH(256L * 128 * 9);
  _Float16* u_fh2_w   = allocH(2L * 256 * 9);
  _Float16* u_mask1_w = allocH(256L * 128 * 9);
  _Float16* u_mask2_w = allocH(576L * 256);
  cvt((const float*)d_in[35], u_conv_w,   126L * 256 * 9);
  cvt((const float*)d_in[37], u_convc1_w, 256L * 36);
  cvt((const float*)d_in[39], u_convc2_w, 192L * 256 * 9);
  cvt((const float*)d_in[41], u_convf1_w, 128L * 2 * 49);
  cvt((const float*)d_in[43], u_convf2_w, 64L * 128 * 9);
  cvt((const float*)d_in[45], u_q1_w, 128L * 384 * 5);
  cvt((const float*)d_in[47], u_q2_w, 128L * 384 * 5);
  cvt((const float*)d_in[49], u_r1_w, 128L * 384 * 5);
  cvt((const float*)d_in[51], u_r2_w, 128L * 384 * 5);
  cvt((const float*)d_in[53], u_z1_w, 128L * 384 * 5);
  cvt((const float*)d_in[55], u_z2_w, 128L * 384 * 5);
  cvt((const float*)d_in[57], u_fh1_w,   256L * 128 * 9);
  cvt((const float*)d_in[59], u_fh2_w,   2L * 256 * 9);
  cvt((const float*)d_in[61], u_mask1_w, 256L * 128 * 9);
  cvt((const float*)d_in[63], u_mask2_w, 576L * 256);

  _Float16* frameh = allocH((long)NB * 3 * H0 * W0);
  float*    c1  = allocF((long)NB * 64  * H1 * W1);
  _Float16* c1h = allocH((long)NB * 64  * H1 * W1);
  float*    c2  = allocF((long)NB * 128 * H2 * W2);
  _Float16* c2h = allocH((long)NB * 128 * H2 * W2);
  float*    c3  = allocF((long)NB * 192 * H3 * W3);
  _Float16* c3h = allocH((long)NB * 192 * H3 * W3);

  float* fmap[2]; _Float16* fmaph[2];
  float* dw8[2];  _Float16* dw8h[2];
  float* dw16[2]; _Float16* dw16h[2];
  for (int f = 0; f < 2; ++f) {
    fmap[f]  = allocF((long)NB * 256 * HWmax);
    fmaph[f] = allocH((long)NB * 256 * HWmax);
    dw8[f]   = allocF((long)NB * 256 * H8 * W8);
    dw8h[f]  = allocH((long)NB * 256 * H8 * W8);
    dw16[f]  = allocF((long)NB * 256 * L16);
    dw16h[f] = allocH((long)NB * 256 * L16);
  }
  float* offs8  = allocF((long)NB * 18 * H8 * W8);
  float* offs16 = allocF((long)NB * 18 * L16);
  float* net   = allocF((long)NB * 128 * HWmax);
  float* inp   = allocF((long)NB * 128 * HWmax);
  float* net8  = allocF((long)NB * 128 * H8 * W8);
  float* inp8  = allocF((long)NB * 128 * H8 * W8);
  float* net16 = allocF((long)NB * 128 * L16);
  float* inp16 = allocF((long)NB * 128 * L16);

  const long AT = (long)NB * 256 * L16;
  float*    f1peF = allocF(AT); _Float16* f1peH = allocH(AT);
  float*    f2peF = allocF(AT); _Float16* f2peH = allocH(AT);
  float*    f1sF  = allocF(AT); _Float16* f1sH  = allocH(AT);
  float*    f2sF  = allocF(AT); _Float16* f2sH  = allocH(AT);
  float*    attlF = allocF(AT); _Float16* attlH = allocH(AT);
  float*    attrF = allocF(AT); _Float16* attrH = allocH(AT);
  float* Qf = allocF(AT);
  float* Kf = allocF(AT);
  float* Vf = allocF(AT);
  float* KVb = allocF((long)NB * 8 * 32 * 32);
  float* Ksb = allocF((long)NB * 8 * 32);
  float* Zb  = allocF((long)NB * 8 * L16);
  _Float16* msgH = allocH(AT);
  float* mrgF  = allocF(AT);
  float* msgnF = allocF(AT);
  _Float16* cat512H = allocH((long)NB * 512 * L16);
  _Float16* mlp1H   = allocH((long)NB * 512 * L16);
  float* ybF = allocF(AT);

  _Float16* corrh  = allocH((long)NB * 36 * HWmax);
  _Float16* flowh  = allocH((long)NB * 2 * HWmax);
  _Float16* mc1h   = allocH((long)NB * 256 * HWmax);
  _Float16* mc2h   = allocH((long)NB * 192 * HWmax);
  _Float16* mf1h   = allocH((long)NB * 128 * HWmax);
  _Float16* mf2h   = allocH((long)NB * 64 * HWmax);
  _Float16* mcath  = allocH((long)NB * 256 * HWmax);
  _Float16* mouth  = allocH((long)NB * 126 * HWmax);
  _Float16* hxh    = allocH((long)NB * 384 * HWmax);
  _Float16* rxh    = allocH((long)NB * 384 * HWmax);
  float* zF = allocF((long)NB * 128 * HWmax);
  float* rF = allocF((long)NB * 128 * HWmax);
  float* qF = allocF((long)NB * 128 * HWmax);
  _Float16* neth   = allocH((long)NB * 128 * HWmax);
  _Float16* fh1H   = allocH((long)NB * 256 * HWmax);
  float* deltaF    = allocF((long)NB * 2 * HWmax);
  _Float16* mask1H = allocH((long)NB * 256 * HWmax);
  float* maskF     = allocF((long)NB * 576 * HWmax);
  float* rsF       = allocF((long)NB * 256 * HWmax);

  float* flow16F = allocF((long)NB * 2 * L16);
  float* flow8F  = allocF((long)NB * 2 * H8 * W8);
  float* flowFF  = allocF((long)NB * 2 * HWmax);
  float* flowupF = allocF((long)NB * 2 * 240 * 320);

  auto launch1 = [&](long n) { return dim3((unsigned)cdiv(n, 256)); };

  cvt(fw1, w1h, 64L * 3 * 49);
  cvt(fw2, w2h, 128L * 64 * 9);
  cvt(fw3, w3h, 192L * 128 * 9);
  cvt(fw4, w4h, 256L * 192);
  cvt(off8_w,  o8wh,  18L * 256 * 9);
  cvt(off16_w, o16wh, 18L * 256 * 9);

  const float* frames[2] = { frame1, frame2 };
  for (int f = 0; f < 2; ++f) {
    cvt(frames[f], frameh, (long)NB * 3 * H0 * W0);
    conv(w1h, frameh, fb1, c1, nullptr, 64, 3, 7, 7, H0, W0, 2, 3, 3, ACT_NONE);
    k_instnorm_relu<<<dim3(NB * 64), 256, 0, stream>>>(c1, c1h, H1 * W1);
    conv(w2h, c1h, fb2, c2, nullptr, 128, 64, 3, 3, H1, W1, 2, 1, 1, ACT_NONE);
    k_instnorm_relu<<<dim3(NB * 128), 256, 0, stream>>>(c2, c2h, H2 * W2);
    conv(w3h, c2h, fb3, c3, nullptr, 192, 128, 3, 3, H2, W2, 2, 1, 1, ACT_NONE);
    k_instnorm_relu<<<dim3(NB * 192), 256, 0, stream>>>(c3, c3h, H3 * W3);
    conv(w4h, c3h, fb4, fmap[f], fmaph[f], 256, 192, 1, 1, H3, W3, 1, 0, 0, ACT_NONE);
    {
      long n8 = (long)NB * 256 * H8 * W8;
      k_avgpool<<<launch1(n8), 256, 0, stream>>>(fmap[f], dw8[f], dw8h[f], NB * 256, H3, W3, 2);
      long n16 = (long)NB * 256 * L16;
      k_avgpool<<<launch1(n16), 256, 0, stream>>>(fmap[f], dw16[f], dw16h[f], NB * 256, H3, W3, 4);
    }
  }

  conv(o8wh,  dw8h[0],  off8_b,  offs8,  nullptr, 18, 256, 3, 3, H8,  W8,  1, 1, 1, ACT_SIGSCALE);
  conv(o16wh, dw16h[0], off16_b, offs16, nullptr, 18, 256, 3, 3, H16, W16, 1, 1, 1, ACT_SIGSCALE);

  k_net_inp<<<launch1((long)NB * 128 * HWmax), 256, 0, stream>>>(fmap[0], net, inp, NB, HWmax);
  k_avgpool<<<launch1((long)NB * 128 * H8 * W8), 256, 0, stream>>>(net, net8,  nullptr, NB * 128, H3, W3, 2);
  k_avgpool<<<launch1((long)NB * 128 * H8 * W8), 256, 0, stream>>>(inp, inp8,  nullptr, NB * 128, H3, W3, 2);
  k_avgpool<<<launch1((long)NB * 128 * L16), 256, 0, stream>>>(net, net16, nullptr, NB * 128, H3, W3, 4);
  k_avgpool<<<launch1((long)NB * 128 * L16), 256, 0, stream>>>(inp, inp16, nullptr, NB * 128, H3, W3, 4);

  auto loftr = [&](const float* xf, const _Float16* xh, const _Float16* sh,
                   float* outF, _Float16* outH, const AttW& aw) {
    const int L = L16;
    conv(aw.qw, xh, nullptr, Qf, nullptr, 256, 256, 1, 1, 1, L, 1, 0, 0, ACT_ELU1);
    conv(aw.kw, sh, nullptr, Kf, nullptr, 256, 256, 1, 1, 1, L, 1, 0, 0, ACT_ELU1);
    conv(aw.vw, sh, nullptr, Vf, nullptr, 256, 256, 1, 1, 1, L, 1, 0, 0, ACT_NONE);
    k_kv_ksum<<<dim3(NB * 8), 256, 0, stream>>>(Kf, Vf, KVb, Ksb, L);
    k_zcoef<<<launch1((long)NB * 8 * L), 256, 0, stream>>>(Qf, Ksb, Zb, NB, L);
    k_msg<<<launch1((long)NB * 256 * L), 256, 0, stream>>>(Qf, KVb, Zb, msgH, NB, L);
    conv(aw.mw, msgH, nullptr, mrgF, nullptr, 256, 256, 1, 1, 1, L, 1, 0, 0, ACT_NONE);
    k_layernorm_fm<<<launch1((long)NB * L), 256, 0, stream>>>(
        mrgF, aw.ln1g, aw.ln1b, nullptr, msgnF, nullptr, NB, 256, L);
    packF(xf,    256, 0, cat512H, 512, 0,   256, L);
    packF(msgnF, 256, 0, cat512H, 512, 256, 256, L);
    conv(aw.m1w, cat512H, nullptr, nullptr, mlp1H, 512, 512, 1, 1, 1, L, 1, 0, 0, ACT_RELU);
    conv(aw.m2w, mlp1H, nullptr, ybF, nullptr, 256, 512, 1, 1, 1, L, 1, 0, 0, ACT_NONE);
    k_layernorm_fm<<<launch1((long)NB * L), 256, 0, stream>>>(
        ybF, aw.ln2g, aw.ln2b, xf, outF, outH, NB, 256, L);
  };

  k_posenc<<<launch1(AT), 256, 0, stream>>>(dw16[0], f1peF, f1peH, NB, H16, W16);
  k_posenc<<<launch1(AT), 256, 0, stream>>>(dw16[1], f2peF, f2peH, NB, H16, W16);
  loftr(f1peF, f1peH, f1peH, f1sF, f1sH, att[0]);
  loftr(f2peF, f2peH, f2peH, f2sF, f2sH, att[0]);

  auto update_block = [&](int H, int Wd, float* netF, const float* inpF,
                          float* flowF, bool do_up, float* up_out) {
    int HW = H * Wd;
    cvt(flowF, flowh, (long)NB * 2 * HW);
    conv(u_convc1_w, corrh, u_convc1_b, nullptr, mc1h, 256, 36, 1, 1, H, Wd, 1, 0, 0, ACT_RELU);
    conv(u_convc2_w, mc1h, u_convc2_b, nullptr, mc2h, 192, 256, 3, 3, H, Wd, 1, 1, 1, ACT_RELU);
    conv(u_convf1_w, flowh, u_convf1_b, nullptr, mf1h, 128, 2, 7, 7, H, Wd, 1, 3, 3, ACT_RELU);
    conv(u_convf2_w, mf1h, u_convf2_b, nullptr, mf2h, 64, 128, 3, 3, H, Wd, 1, 1, 1, ACT_RELU);
    packH(mc2h, 192, 0, mcath, 256, 0,   192, HW);
    packH(mf2h,  64, 0, mcath, 256, 192,  64, HW);
    conv(u_conv_w, mcath, u_conv_b, nullptr, mouth, 126, 256, 3, 3, H, Wd, 1, 1, 1, ACT_RELU);
    packF(netF, 128, 0, hxh, 384, 0, 128, HW);
    packF(inpF, 128, 0, hxh, 384, 128, 128, HW);
    packH(mouth, 126, 0, hxh, 384, 256, 126, HW);
    packH(flowh, 2, 0, hxh, 384, 382, 2, HW);
    conv(u_z1_w, hxh, u_z1_b, zF, nullptr, 128, 384, 1, 5, H, Wd, 1, 0, 2, ACT_SIGMOID);
    conv(u_r1_w, hxh, u_r1_b, rF, nullptr, 128, 384, 1, 5, H, Wd, 1, 0, 2, ACT_SIGMOID);
    k_pack_rh<<<launch1((long)NB * 128 * HW), 256, 0, stream>>>(rF, netF, rxh, NB, HW);
    packH(hxh, 384, 128, rxh, 384, 128, 256, HW);
    conv(u_q1_w, rxh, u_q1_b, qF, nullptr, 128, 384, 1, 5, H, Wd, 1, 0, 2, ACT_TANH);
    k_gru_update<<<launch1((long)NB * 128 * HW), 256, 0, stream>>>(netF, zF, qF, (long)NB * 128 * HW);
    packF(netF, 128, 0, hxh, 384, 0, 128, HW);
    conv(u_z2_w, hxh, u_z2_b, zF, nullptr, 128, 384, 5, 1, H, Wd, 1, 2, 0, ACT_SIGMOID);
    conv(u_r2_w, hxh, u_r2_b, rF, nullptr, 128, 384, 5, 1, H, Wd, 1, 2, 0, ACT_SIGMOID);
    k_pack_rh<<<launch1((long)NB * 128 * HW), 256, 0, stream>>>(rF, netF, rxh, NB, HW);
    conv(u_q2_w, rxh, u_q2_b, qF, nullptr, 128, 384, 5, 1, H, Wd, 1, 2, 0, ACT_TANH);
    k_gru_update<<<launch1((long)NB * 128 * HW), 256, 0, stream>>>(netF, zF, qF, (long)NB * 128 * HW);
    packF(netF, 128, 0, neth, 128, 0, 128, HW);
    conv(u_fh1_w, neth, u_fh1_b, nullptr, fh1H, 256, 128, 3, 3, H, Wd, 1, 1, 1, ACT_RELU);
    conv(u_fh2_w, fh1H, u_fh2_b, deltaF, nullptr, 2, 256, 3, 3, H, Wd, 1, 1, 1, ACT_NONE);
    k_add_inplace<<<launch1((long)NB * 2 * HW), 256, 0, stream>>>(flowF, deltaF, (long)NB * 2 * HW);
    if (do_up) {
      conv(u_mask1_w, neth, u_mask1_b, nullptr, mask1H, 256, 128, 3, 3, H, Wd, 1, 1, 1, ACT_RELU);
      conv(u_mask2_w, mask1H, u_mask2_b, maskF, nullptr, 576, 256, 1, 1, H, Wd, 1, 0, 0, ACT_QUARTER);
      k_upsample<<<launch1((long)NB * HW * 64), 256, 0, stream>>>(flowF, maskF, up_out, NB, H, Wd);
    }
  };

  k_zero_f32<<<launch1((long)NB * 2 * L16), 256, 0, stream>>>(flow16F, (long)NB * 2 * L16);
  for (int itr = 0; itr < 2; ++itr) {
    bool small = (itr % 2 == 1);
    int rx = small ? 1 : 4, ry = small ? 1 : 0;
    loftr(f1sF, f1sH, f2sH, attlF, attlH, att[1]);
    loftr(f2sF, f2sH, attlH, attrF, attrH, att[1]);
    k_corr_offset<<<launch1((long)NB * 9 * L16), 256, 0, stream>>>(
        attlF, attrF, flow16F, offs16, corrh, NB, H16, W16, rx, ry);
    update_block(H16, W16, net16, inp16, flow16F, itr == 1, flowupF);
  }
  k_interp<<<launch1((long)NB * 2 * H8 * W8), 256, 0, stream>>>(
      flowupF, flow8F, NB, 2, 120, 160, H8, W8, 0.25f);

  for (int itr = 0; itr < 2; ++itr) {
    bool small = (itr % 2 == 1);
    int rx = small ? 1 : 4, ry = small ? 1 : 0;
    k_corr_offset<<<launch1((long)NB * 9 * H8 * W8), 256, 0, stream>>>(
        dw8[0], dw8[1], flow8F, offs8, corrh, NB, H8, W8, rx, ry);
    update_block(H8, W8, net8, inp8, flow8F, itr == 1, flowupF);
  }
  k_interp<<<launch1((long)NB * 2 * HWmax), 256, 0, stream>>>(
      flowupF, flowFF, NB, 2, 240, 320, H3, W3, 0.25f);

  for (int itr = 0; itr < 4; ++itr) {
    k_bilin_all<<<launch1((long)NB * 256 * HWmax), 256, 0, stream>>>(
        fmap[1], flowFF, rsF, NB, H3, W3);
    bool small = (itr % 2 == 1);
    int rx = small ? 1 : 4, ry = small ? 1 : 0;
    k_corr_shift<<<launch1((long)NB * 9 * HWmax), 256, 0, stream>>>(
        fmap[0], rsF, corrh, NB, H3, W3, rx, ry);
    update_block(H3, W3, net, inp, flowFF, itr == 3, (float*)d_out);
  }
}
